// BiLSTM_CRF_11321533792557
// MI455X (gfx1250) — matrix operand order repaired and checked by reading, not compiled, not run on hardware
//
#include <hip/hip_runtime.h>
#include <hip/hip_bf16.h>

// ---------------------------------------------------------------------------
// BiLSTM-CRF for MI455X (gfx1250, wave32).
//   T=4096, LC=20, CV=100, CE=25, CO=25, V=100000, E=512, H=1024, NT=12
// Pipeline:
//   k_init       : zero h double-buffers + grid-barrier counter (graph-safe)
//   k_embed      : char-CNN + word-emb gather -> emb bf16 [T][544] (K padded)
//   k_prep_*     : w_ih->bf16 padded, bias fuse, W_hh per-WG packing
//   k_gemm_wmma  : pre[dir][t][4H] = emb @ w_ih.T + b   (v_wmma_f32_16x16x32_bf16)
//   k_lstm       : sequential recurrence distributed over 16 WGs/direction;
//                  each WG owns 64 cells (all 4 gates -> no gate exchange),
//                  h exchanged via L2 double-buffer + device atomic barrier.
//                  Per WG/step: 1MB coalesced b128 weight stream (16 WGPs pull
//                  the 16MB W_hh in parallel from the 192MB L2).
//   k_feats      : [T,2H] @ w_tag.T + b_tag
//   k_viterbi    : 12-state CRF forward DP + backtrace (1 wave)
// ---------------------------------------------------------------------------

#define T_LEN 4096
#define KP 544          // 537 padded to 17*32
#define HX 1024
#define FH 4096         // 4*H
#define NT 12
#define TAG_START 10
#define TAG_STOP 11
#define NEGV (-10000.0f)
#define NWG 16          // workgroups per direction in k_lstm
#define NBLK 32         // total persistent blocks (2 dirs * NWG)

typedef __attribute__((ext_vector_type(16))) __bf16 v16bf;
typedef __attribute__((ext_vector_type(8)))  float  v8f;

__device__ __forceinline__ unsigned short f32_to_bf16(float f) {
  unsigned int u = __float_as_uint(f);
  unsigned int r = u + 0x7FFFu + ((u >> 16) & 1u);   // RNE
  return (unsigned short)(r >> 16);
}
__device__ __forceinline__ float sigmoidf(float x) {
  return 1.0f / (1.0f + __expf(-x));
}

// ---------------- per-call init (graph replay safe) -------------------------
__global__ void k_init(float* __restrict__ hbuf /* [2][2][HX] */,
                       int* __restrict__ bar) {
  int i = blockIdx.x * 256 + threadIdx.x;
  if (i < 2 * 2 * HX) hbuf[i] = 0.f;
  if (i == 0) *bar = 0;
}

// ---------------- char CNN + embedding -> bf16 padded rows ------------------
__global__ __launch_bounds__(128)
void k_embed(const int* __restrict__ sentence, const int* __restrict__ chars,
             const float* __restrict__ char_emb, const float* __restrict__ cnn_w,
             const float* __restrict__ cnn_b, const float* __restrict__ word_emb,
             unsigned short* __restrict__ embB /* [T][KP] */) {
  int t = blockIdx.x;
  int tid = threadIdx.x;
  __shared__ float ce[20 * 25];
  for (int idx = tid; idx < 20 * 25; idx += 128) {
    int l = idx / 25, e = idx % 25;
    int ch = chars[t * 20 + l];
    ce[idx] = char_emb[ch * 25 + e];
  }
  int w = sentence[t];
  const float* wr = word_emb + (size_t)w * 512;
  unsigned short* er = embB + (size_t)t * KP;
  for (int k = tid; k < 512; k += 128) er[k] = f32_to_bf16(wr[k]);
  if (tid < KP - 537) er[537 + tid] = 0;              // zero K padding
  __syncthreads();
  if (tid < 25) {                                     // one output channel each
    int co = tid;
    float best = -3.4e38f;
    for (int h = 0; h < 22; ++h) {                    // LC + 2 output rows
      float s = 0.f;
      #pragma unroll
      for (int r = 0; r < 3; ++r) {
        int row = h + r - 2;
        if (row < 0 || row >= 20) continue;
        const float* kw = cnn_w + (co * 3 + r) * 25;
        const float* cr = ce + row * 25;
        #pragma unroll
        for (int e = 0; e < 25; ++e) s += cr[e] * kw[e];
      }
      best = (s > best) ? s : best;
    }
    er[512 + co] = f32_to_bf16(best + cnn_b[co]);
  }
}

// ---------------- weight prep ----------------------------------------------
__global__ void k_prep_wih(const float* __restrict__ wf, const float* __restrict__ wb,
                           unsigned short* __restrict__ out /* [2][FH][KP] */) {
  size_t i = (size_t)blockIdx.x * 256 + threadIdx.x;
  const size_t tot = 2ull * FH * KP;
  if (i >= tot) return;
  int dir = (int)(i / ((size_t)FH * KP));
  size_t rem = i % ((size_t)FH * KP);
  int n = (int)(rem / KP), k = (int)(rem % KP);
  const float* w = dir ? wb : wf;
  out[i] = (k < 537) ? f32_to_bf16(w[(size_t)n * 537 + k]) : (unsigned short)0;
}

__global__ void k_prep_bias(const float* bihf, const float* bhhf,
                            const float* bihb, const float* bhhb,
                            float* __restrict__ bias /* [2][FH] */) {
  int i = blockIdx.x * 256 + threadIdx.x;
  if (i >= 2 * FH) return;
  int dir = i >> 12, n = i & (FH - 1);
  bias[i] = dir ? (bihb[n] + bhhb[n]) : (bihf[n] + bhhf[n]);
}

// Pack W_hh into per-WG streaming layout:
//   wpack[((dir*NWG + w)*256 + k4)*256 + tcol][4]  (float4 per lane per k4)
// where tcol = g*64 + u -> output column j = g*1024 + 64*w + u, k = 4*k4 + s.
__global__ void k_prep_wpack(const float* __restrict__ wf, const float* __restrict__ wb,
                             float* __restrict__ out) {
  size_t i = (size_t)blockIdx.x * 256 + threadIdx.x;
  const size_t tot = 2ull * NWG * 256 * 256 * 4;      // 8,388,608 = 32MB
  if (i >= tot) return;
  int s = (int)(i & 3);
  size_t r = i >> 2;
  int tcol = (int)(r & 255); r >>= 8;
  int k4   = (int)(r & 255); r >>= 8;
  int w    = (int)(r & (NWG - 1)); r >>= 4;
  int dir  = (int)r;
  int g = tcol >> 6, u = tcol & 63;
  int j = g * 1024 + 64 * w + u;
  int k = 4 * k4 + s;
  const float* src = dir ? wb : wf;
  out[i] = src[(size_t)j * HX + k];
}

// ---------------- input projection GEMM on the WMMA pipes -------------------
// pre[dir][m][n] = sum_k emb[dir? T-1-m : m][k] * w_ih[dir][n][k]  + bias
__global__ __launch_bounds__(256)
void k_gemm_wmma(const unsigned short* __restrict__ embB,   // [T][KP] bf16
                 const unsigned short* __restrict__ wihB,   // [2][FH][KP] bf16
                 const float* __restrict__ bias,            // [2][FH]
                 float* __restrict__ pre) {                  // [2][T][FH]
  int dir  = blockIdx.z;
  int m0   = blockIdx.x * 16;
  int wave = threadIdx.x >> 5;
  int lane = threadIdx.x & 31;
  int n0   = blockIdx.y * 128 + wave * 16;
  int half = lane >> 4;       // lane group
  int l15  = lane & 15;

  int mrow = m0 + l15;
  int arow = dir ? (T_LEN - 1 - mrow) : mrow;               // reversed scan for bwd
  // A frag (16x32 bf16): lanes0-15 hold K{0-7,16-23}; lanes16-31 hold K{8-15,24-31}
  const unsigned short* aptr = embB + (size_t)arow * KP + (half ? 8 : 0);
  // B frag (32x16 bf16): lanes0-15 hold K{0-7,16-23}; lanes16-31 hold K{8-15,24-31} of col n0+l15
  const unsigned short* bptr = wihB + ((size_t)dir * FH + (n0 + l15)) * KP + (half ? 8 : 0);

  union Frag { v16bf v; uint4 q[2]; };
  v8f acc = {0.f, 0.f, 0.f, 0.f, 0.f, 0.f, 0.f, 0.f};
  for (int kt = 0; kt < 17; ++kt) {
    Frag a, b;
    const unsigned short* ap = aptr + kt * 32;
    const unsigned short* bp = bptr + kt * 32;
    a.q[0] = *(const uint4*)(ap);
    a.q[1] = *(const uint4*)(ap + 16);
    b.q[0] = *(const uint4*)(bp);
    b.q[1] = *(const uint4*)(bp + 16);
    if (kt + 1 < 17) {                                      // global_prefetch_b8
      __builtin_prefetch((const void*)(ap + 32), 0, 0);
      __builtin_prefetch((const void*)(bp + 32), 0, 0);
    }
    acc = __builtin_amdgcn_wmma_f32_16x16x32_bf16(
        false, a.v, false, b.v, (short)0, acc, false, false);
  }
  float bn = bias[dir * FH + n0 + l15];
  // C/D layout: VGPR r -> row m0 + r + 8*half, col n0 + l15
  #pragma unroll
  for (int r = 0; r < 8; ++r) {
    int row = m0 + r + half * 8;
    pre[((size_t)dir * T_LEN + row) * FH + (n0 + l15)] = acc[r] + bn;
  }
}

// ---------------- distributed sequential LSTM recurrence --------------------
// 32 persistent blocks (2 dirs x 16 WGs), 256 threads each.
// Block (dir, w) owns cells [64w, 64w+64) and computes its 256 GEMV columns
// {g*1024 + 64w + u : g in 0..3, u in 0..63} per step. h is double-buffered in
// L2; one device-scope atomic barrier per step.
__global__ __launch_bounds__(256)
void k_lstm(const float* __restrict__ pre,    // [2][T][FH] (bwd already reversed)
            const float* __restrict__ wpack,  // [2][NWG][256][256][4]
            float* __restrict__ hout,         // [2][T][HX] (natural time order)
            float* __restrict__ hbuf,         // [2][2][HX] step-parity buffers
            int* __restrict__ bar) {
  int blk = blockIdx.x;
  int dir = blk >> 4;
  int w   = blk & (NWG - 1);
  int tid = threadIdx.x;
  int u = tid & 63, g = tid >> 6;
  int jcol = g * 1024 + 64 * w + u;

  __shared__ float lh[HX];     // current h, broadcast-read by the k-loop
  __shared__ float ly[256];    // this WG's 256 gate pre-activations

  const float* wp   = wpack + ((size_t)(dir * NWG + w) * 256 * 256 * 4) + tid * 4;
  const float* preD = pre + (size_t)dir * T_LEN * FH;
  float* hD  = hout + (size_t)dir * T_LEN * HX;
  float* hb0 = hbuf + dir * 2 * HX;
  float c = 0.f;               // cell state for tid<64

  for (int t = 0; t < T_LEN; ++t) {
    // stage h(t) into LDS (4KB, L2-resident, shared by all 16 WGs)
    {
      float4 hv = *(const float4*)(hb0 + (t & 1) * HX + 4 * tid);
      *(float4*)(lh + 4 * tid) = hv;
    }
    __syncthreads();
    // 256-col slice of y = h . W_hh^T : one b128 weight load per lane per k4
    float acc = 0.f;
    #pragma unroll 4
    for (int k4 = 0; k4 < 256; ++k4) {
      float4 wv = *(const float4*)(wp + (size_t)k4 * 1024);
      float4 hh = *(const float4*)(&lh[4 * k4]);
      acc = fmaf(hh.x, wv.x, acc);
      acc = fmaf(hh.y, wv.y, acc);
      acc = fmaf(hh.z, wv.z, acc);
      acc = fmaf(hh.w, wv.w, acc);
    }
    acc += preD[(size_t)t * FH + jcol];
    ly[tid] = acc;
    __syncthreads();
    if (tid < 64) {                          // own cells: all 4 gates local
      float ig = sigmoidf(ly[tid]);
      float fg = sigmoidf(ly[64 + tid]);
      float gg = tanhf(ly[128 + tid]);
      float og = sigmoidf(ly[192 + tid]);
      float cv = fg * c + ig * gg;
      c = cv;
      float hv = og * tanhf(cv);
      int cell = 64 * w + tid;
      int trow = dir ? (T_LEN - 1 - t) : t;
      hD[(size_t)trow * HX + cell] = hv;
      hb0[((t + 1) & 1) * HX + cell] = hv;   // publish for next step
    }
    // device-scope barrier across all 32 blocks (monotonic counter)
    __threadfence();
    __syncthreads();
    if (tid == 0) {
      __hip_atomic_fetch_add(bar, 1, __ATOMIC_RELEASE, __HIP_MEMORY_SCOPE_AGENT);
      int target = NBLK * (t + 1);
      while (__hip_atomic_load(bar, __ATOMIC_ACQUIRE, __HIP_MEMORY_SCOPE_AGENT) < target)
        __builtin_amdgcn_s_sleep(2);
    }
    __syncthreads();
  }
}

// ---------------- tag projection -------------------------------------------
__global__ __launch_bounds__(256)
void k_feats(const float* __restrict__ hout, const float* __restrict__ w_tag,
             const float* __restrict__ b_tag, float* __restrict__ feats) {
  int t = blockIdx.x;
  int tid = threadIdx.x, lane = tid & 31, wid = tid >> 5;
  const float* hf = hout + (size_t)t * HX;
  const float* hb = hout + ((size_t)T_LEN + t) * HX;
  float acc[NT];
  #pragma unroll
  for (int n = 0; n < NT; ++n) acc[n] = 0.f;
  for (int k = tid; k < HX; k += 256) {
    float x = hf[k], y = hb[k];
    #pragma unroll
    for (int n = 0; n < NT; ++n)
      acc[n] += x * w_tag[n * 2048 + k] + y * w_tag[n * 2048 + HX + k];
  }
  __shared__ float part[8][NT];
  #pragma unroll
  for (int n = 0; n < NT; ++n) {
    float v = acc[n];
    for (int off = 16; off > 0; off >>= 1) v += __shfl_down(v, off, 32);
    if (lane == 0) part[wid][n] = v;
  }
  __syncthreads();
  if (tid < NT) {
    float s = b_tag[tid];
    #pragma unroll
    for (int w = 0; w < 8; ++w) s += part[w][tid];
    feats[(size_t)t * NT + tid] = s;
  }
}

// ---------------- Viterbi / CRF forward DP + backtrace (1 wave) -------------
__global__ __launch_bounds__(32)
void k_viterbi(const float* __restrict__ feats, const float* __restrict__ trans,
               int* __restrict__ bptrs, float* __restrict__ out) {
  __shared__ float la[NT];
  int j = threadIdx.x;
  if (j < NT) la[j] = (j == TAG_START) ? 0.f : NEGV;
  __syncthreads();
  for (int t = 0; t < T_LEN; ++t) {
    float na = 0.f; int bp = 0;
    if (j < NT) {
      float ft = feats[(size_t)t * NT + j];
      float v[NT], m = -3.4e38f; int am = 0;
      #pragma unroll
      for (int i = 0; i < NT; ++i) {
        float s = la[i] + trans[i * NT + j];
        v[i] = s;
        if (s > m) { m = s; am = i; }
      }
      float sum = 0.f;
      #pragma unroll
      for (int i = 0; i < NT; ++i) sum += __expf(v[i] - m);
      na = ft + m + __logf(sum);
      bp = am;
    }
    __syncthreads();
    if (j < NT) { la[j] = na; bptrs[(size_t)t * NT + j] = bp; }
    __syncthreads();
  }
  if (j == 0) {
    float m = -3.4e38f; int best = 0;
    float fin[NT];
    for (int i = 0; i < NT; ++i) {
      fin[i] = la[i] + trans[i * NT + TAG_STOP];
      if (fin[i] > m) { m = fin[i]; best = i; }
    }
    float sum = 0.f;
    for (int i = 0; i < NT; ++i) sum += __expf(fin[i] - m);
    out[0] = m + __logf(sum);                     // score
    out[1 + T_LEN - 1] = (float)best;             // path (floats)
    int cur = best;
    for (int t = T_LEN - 1; t >= 1; --t) {
      cur = bptrs[(size_t)t * NT + cur];
      out[1 + t - 1] = (float)cur;
    }
  }
}

// ---------------------------------------------------------------------------
extern "C" void kernel_launch(void* const* d_in, const int* in_sizes, int n_in,
                              void* d_out, int out_size, void* d_ws, size_t ws_size,
                              hipStream_t stream) {
  const int*   sentence = (const int*)  d_in[0];
  const int*   chars    = (const int*)  d_in[1];
  /* caps d_in[2] unused by the reference */
  const float* char_emb = (const float*)d_in[3];
  const float* cnn_w    = (const float*)d_in[4];
  const float* cnn_b    = (const float*)d_in[5];
  const float* word_emb = (const float*)d_in[6];
  const float* w_ih_f   = (const float*)d_in[7];
  const float* w_hh_f   = (const float*)d_in[8];
  const float* b_ih_f   = (const float*)d_in[9];
  const float* b_hh_f   = (const float*)d_in[10];
  const float* w_ih_b   = (const float*)d_in[11];
  const float* w_hh_b   = (const float*)d_in[12];
  const float* b_ih_b   = (const float*)d_in[13];
  const float* b_hh_b   = (const float*)d_in[14];
  const float* w_tag    = (const float*)d_in[15];
  const float* b_tag    = (const float*)d_in[16];
  const float* trans    = (const float*)d_in[17];

  // workspace carve-up (~215 MB total)
  char* ws = (char*)d_ws;
  size_t off = 0;
  auto carve = [&](size_t bytes) -> void* {
    void* p = ws + off;
    off += (bytes + 255) & ~(size_t)255;
    return p;
  };
  unsigned short* embB = (unsigned short*)carve((size_t)T_LEN * KP * 2);
  unsigned short* wihB = (unsigned short*)carve(2ull * FH * KP * 2);
  float* bias  = (float*)carve(2ull * FH * 4);
  float* wpack = (float*)carve(2ull * NWG * 256 * 256 * 4 * 4);
  float* pre   = (float*)carve(2ull * T_LEN * FH * 4);
  float* hout  = (float*)carve(2ull * T_LEN * HX * 4);
  float* feats = (float*)carve((size_t)T_LEN * NT * 4);
  int*   bptrs = (int*)  carve((size_t)T_LEN * NT * 4);
  float* hbuf  = (float*)carve(2ull * 2 * HX * 4);
  int*   bar   = (int*)  carve(256);

  k_init<<<16, 256, 0, stream>>>(hbuf, bar);
  k_embed<<<T_LEN, 128, 0, stream>>>(sentence, chars, char_emb, cnn_w, cnn_b,
                                     word_emb, embB);
  {
    size_t tot = 2ull * FH * KP;
    k_prep_wih<<<(unsigned)((tot + 255) / 256), 256, 0, stream>>>(w_ih_f, w_ih_b, wihB);
  }
  k_prep_bias<<<(2 * FH + 255) / 256, 256, 0, stream>>>(b_ih_f, b_hh_f, b_ih_b,
                                                        b_hh_b, bias);
  {
    size_t tot = 2ull * NWG * 256 * 256 * 4;
    k_prep_wpack<<<(unsigned)((tot + 255) / 256), 256, 0, stream>>>(w_hh_f, w_hh_b, wpack);
  }
  dim3 gg(T_LEN / 16, FH / 128, 2);
  k_gemm_wmma<<<gg, 256, 0, stream>>>(embB, wihB, bias, pre);
  k_lstm<<<NBLK, 256, 0, stream>>>(pre, wpack, hout, hbuf, bar);
  k_feats<<<T_LEN, 256, 0, stream>>>(hout, w_tag, b_tag, feats);
  k_viterbi<<<1, 32, 0, stream>>>(feats, trans, bptrs, (float*)d_out);
}
